// FRAP_61761629716929
// MI455X (gfx1250) — compile-verified, matrix operand order checked by reading, not run on hardware
//
#include <hip/hip_runtime.h>

typedef __attribute__((ext_vector_type(16))) _Float16 v16h;
typedef __attribute__((ext_vector_type(8)))  _Float16 v8h;
typedef __attribute__((ext_vector_type(2)))  _Float16 v2h;
typedef __attribute__((ext_vector_type(8)))  float    v8f;

#define N_MOV   8
#define N_PH    8
#define CH      20
#define CHP     32           // channels padded to 2 WMMA N-tiles
#define LANE_CH 16
#define BPW     32           // batch elements per workgroup
#define NTHREADS 256
#define NWAVES   8
#define ROWS_PER_WG   (BPW * 56)            // 1792
#define TILES_PER_WG  (ROWS_PER_WG / 16)    // 112
#define TILES_PER_WAVE (TILES_PER_WG / NWAVES) // 14

__device__ __forceinline__ float sigmoidf(float x) {
    return 1.0f / (1.0f + __expf(-x));
}

__global__ __launch_bounds__(NTHREADS)
void frap_wmma_kernel(const float* __restrict__ states,
                      const float* __restrict__ p_w,
                      const float* __restrict__ d_w,
                      const float* __restrict__ d_b,
                      const float* __restrict__ lane_w,
                      const float* __restrict__ lane_b,
                      const float* __restrict__ lc_w,
                      const float* __restrict__ lc_b,
                      const float* __restrict__ rel_w,
                      const float* __restrict__ rc_w,
                      const float* __restrict__ rc_b,
                      const float* __restrict__ hid_w,
                      const float* __restrict__ hid_b,
                      const float* __restrict__ bm_w,
                      const float* __restrict__ bm_b,
                      const int*   __restrict__ phase_pairs,
                      const int*   __restrict__ comp_mask,
                      float* __restrict__ out)
{
    __shared__ float relL[N_PH * 7 * CHP];                         // 7168 B, rel gate
    __shared__ float laneL[BPW * N_MOV * LANE_CH];                 // 16384 B, rolled lane MLP
    __shared__ __align__(16) _Float16 pfeat[BPW * N_PH * LANE_CH]; // 8192 B, pair features f16
    __shared__ __align__(16) _Float16 xs[NWAVES * 16 * CHP];       // 8192 B, D->A transpose
    __shared__ float qAcc[BPW * N_PH];                             // 1024 B, output accumulator

    const int tid  = threadIdx.x;
    const int wave = tid >> 5;
    const int lane = tid & 31;
    const int m    = lane & 15;   // M-row (A) / N-col (B,C,D) index within fragment
    const int hi   = lane >> 4;   // half-wave selector per ISA fragment layout
    const int b0   = blockIdx.x * BPW;

    // ---- Stage 0a: zero per-(b,p) accumulator (256 == BPW*N_PH) ----
    qAcc[tid] = 0.0f;

    // ---- Stage 0b: rel gate table, relL[p][q][ch] (padded ch -> 0) ----
    for (int idx = tid; idx < N_PH * 7 * CHP; idx += NTHREADS) {
        const int ch = idx & (CHP - 1);
        const int pq = idx >> 5;           // p*7 + q
        float v = 0.0f;
        if (ch < CH) {
            const int cm = comp_mask[pq];  // 0 or 1
            float acc = rc_b[ch];
#pragma unroll
            for (int e = 0; e < 4; ++e) {
                const float rw = fmaxf(rel_w[cm * 4 + e], 0.0f);
                acc += rw * rc_w[ch * 4 + e];
            }
            v = fmaxf(acc, 0.0f);
        }
        relL[idx] = v;
    }

    // ---- Per-lane WMMA B fragments.
    // A/B 16-bit 32x16 fragment: lane = N col, element e holds K position
    //   pos(e) = hi*8 + (e&7) + (e>=8 ? 16 : 0).
    // lc GEMM uses natural K order. hid GEMM uses an interleaved K permutation
    // (channel ch -> pos 2*ch if ch<16 else 2*(ch-16)+1) so the D->A scratch
    // can be written as packed 32-bit stores; B reads the inverse mapping. ----
    v16h bLc0, bLc1, bHid0, bHid1;
#pragma unroll
    for (int e = 0; e < 16; ++e) {
        const int k = hi * 8 + (e & 7) + ((e >= 8) ? 16 : 0);
        bLc0[e] = (_Float16)lc_w[m * 32 + k];                    // ch m (<16<20)
        bLc1[e] = (m + 16 < CH) ? (_Float16)lc_w[(m + 16) * 32 + k] : (_Float16)0.f;
        // inverse of the interleave: even pos -> ch pos/2, odd pos -> 16 + pos/2
        const int ch = (k & 1) ? (16 + (k >> 1)) : (k >> 1);
        bHid0[e] = (ch < CH) ? (_Float16)hid_w[m * CH + ch] : (_Float16)0.f;
        bHid1[e] = (m + 16 < CH && ch < CH) ? (_Float16)hid_w[(m + 16) * CH + ch]
                                            : (_Float16)0.f;
    }
    const float lcb0 = lc_b[m];
    const float lcb1 = (m + 16 < CH) ? lc_b[m + 16] : 0.0f;
    const float hb0  = hid_b[m];
    const float hb1  = (m + 16 < CH) ? hid_b[m + 16] : 0.0f;
    const float bw0  = bm_w[m];
    const float bw1  = (m + 16 < CH) ? bm_w[m + 16] : 0.0f;

    // ---- Stage 1: lane MLP, one thread per (batch-local, movement) ----
    {
        const int bl = tid >> 3;
        const int mv = tid & 7;
        const float* srow = states + (size_t)(b0 + bl) * 9;
        const int   act = (int)srow[0];
        const float dem = srow[1 + mv];
        const int s0 = phase_pairs[act * 2 + 0];
        const int s1 = phase_pairs[act * 2 + 1];
        const int ext = (mv == s0 || mv == s1) ? 1 : 0;
        float pd[8];
#pragma unroll
        for (int o = 0; o < 4; ++o) pd[o] = sigmoidf(p_w[ext * 4 + o]);
#pragma unroll
        for (int o = 0; o < 4; ++o) pd[4 + o] = sigmoidf(dem * d_w[o] + d_b[o]);
        const int mroll = (mv + 6) & 7;    // rolled position: rolled[m'] = lane[(m'+2)%8]
        float* dst = laneL + (bl * N_MOV + mroll) * LANE_CH;
#pragma unroll
        for (int ch = 0; ch < LANE_CH; ++ch) {
            float a = lane_b[ch];
#pragma unroll
            for (int i = 0; i < 8; ++i) a += pd[i] * lane_w[ch * 8 + i];
            dst[ch] = fmaxf(a, 0.0f);
        }
    }
    __syncthreads();

    // ---- Stage 1b: pair features -> f16 ----
    for (int idx = tid; idx < BPW * N_PH * LANE_CH; idx += NTHREADS) {
        const int ch = idx & 15;
        const int p  = (idx >> 4) & 7;
        const int bl = idx >> 7;
        const int i0 = phase_pairs[p * 2 + 0];
        const int i1 = phase_pairs[p * 2 + 1];
        const float v = laneL[(bl * N_MOV + i0) * LANE_CH + ch] +
                        laneL[(bl * N_MOV + i1) * LANE_CH + ch];
        pfeat[idx] = (_Float16)v;
    }
    __syncthreads();

    // ---- Stage 2: WMMA over 16-row tiles of (b,p,q) rows ----
    _Float16* xsw = xs + wave * 16 * CHP;
    const int t0 = wave * TILES_PER_WAVE;
    for (int t = t0; t < t0 + TILES_PER_WAVE; ++t) {
        // A fragment (16x32 f16): lane m holds row t*16+m; hi selects K 0-7/16-23 vs 8-15/24-31
        v16h a;
        {
            const int row = t * 16 + m;
            const int bl  = row / 56;
            const int pq  = row - bl * 56;
            const int p   = pq / 7;
            const int q   = pq - p * 7;
            const int j   = q + ((q >= p) ? 1 : 0);
            const _Float16* pfb = pfeat + bl * N_PH * LANE_CH;
            const v8h lo  = *(const v8h*)(pfb + p * LANE_CH + hi * 8); // K  0..15 slice
            const v8h hi8 = *(const v8h*)(pfb + j * LANE_CH + hi * 8); // K 16..31 slice
#pragma unroll
            for (int e = 0; e < 8; ++e) { a[e] = lo[e]; a[8 + e] = hi8[e]; }
        }

        // D-row decode, shared by rel-scale and final accumulation phases.
        // D layout: VGPR r, half hi  ->  M = r + 8*hi.
        int blR[8], pqR[8], pR[8];
#pragma unroll
        for (int r = 0; r < 8; ++r) {
            const int row = t * 16 + 8 * hi + r;
            blR[r] = row / 56;
            pqR[r] = row - blR[r] * 56;
            pR[r]  = pqR[r] / 7;
        }

        // lc GEMM: 32 -> 32(padded), bias via C-init
        v8f c0, c1;
#pragma unroll
        for (int r = 0; r < 8; ++r) { c0[r] = lcb0; c1[r] = lcb1; }
        v8f x0 = __builtin_amdgcn_wmma_f32_16x16x32_f16(false, a, false, bLc0,
                                                        (short)0, c0, false, false);
        v8f x1 = __builtin_amdgcn_wmma_f32_16x16x32_f16(false, a, false, bLc1,
                                                        (short)0, c1, false, false);

        // relu * rel, packed f16 pair per row -> one 32-bit LDS store per r
        // (interleaved K: ch m -> pos 2m, ch 16+m -> pos 2m+1)
#pragma unroll
        for (int r = 0; r < 8; ++r) {
            const int base = pqR[r] * CHP;
            const float v0 = fmaxf(x0[r], 0.0f) * relL[base + m];
            const float v1 = fmaxf(x1[r], 0.0f) * relL[base + 16 + m];
            v2h pk; pk[0] = (_Float16)v0; pk[1] = (_Float16)v1;
            *(v2h*)(xsw + (r + 8 * hi) * CHP + 2 * m) = pk;
        }

        // hid A fragment from scratch (same-wave LDS; positions are contiguous
        // in the interleaved layout, so plain b128 loads)
        v16h ah;
        {
            const v8h lo  = *(const v8h*)(xsw + m * CHP + hi * 8);
            const v8h hi8 = *(const v8h*)(xsw + m * CHP + 16 + hi * 8);
#pragma unroll
            for (int e = 0; e < 8; ++e) { ah[e] = lo[e]; ah[8 + e] = hi8[e]; }
        }

        v8f h0c, h1c;
#pragma unroll
        for (int r = 0; r < 8; ++r) { h0c[r] = hb0; h1c[r] = hb1; }
        v8f h0 = __builtin_amdgcn_wmma_f32_16x16x32_f16(false, ah, false, bHid0,
                                                        (short)0, h0c, false, false);
        v8f h1 = __builtin_amdgcn_wmma_f32_16x16x32_f16(false, ah, false, bHid1,
                                                        (short)0, h1c, false, false);

        // bm dot: per-lane scale then 16-lane tree reduction (channels live across lanes)
        v8f rowAcc;
#pragma unroll
        for (int r = 0; r < 8; ++r)
            rowAcc[r] = fmaxf(h0[r], 0.0f) * bw0 + fmaxf(h1[r], 0.0f) * bw1;
#pragma unroll
        for (int off = 8; off >= 1; off >>= 1) {
#pragma unroll
            for (int r = 0; r < 8; ++r)
                rowAcc[r] += __shfl_xor(rowAcc[r], off, 16);
        }

        if (m == 0) {  // lanes 0 and 16 hold the group sums for rows r+8*hi
#pragma unroll
            for (int r = 0; r < 8; ++r)
                atomicAdd(&qAcc[blR[r] * N_PH + pR[r]], rowAcc[r]);
        }
    }

    __syncthreads();
    // q[b,p] = sum_q s + 7*bm_b  (bm_b added once per (p,q) in the reference)
    out[(size_t)b0 * N_PH + tid] = qAcc[tid] + 7.0f * bm_b[0];
}

extern "C" void kernel_launch(void* const* d_in, const int* in_sizes, int n_in,
                              void* d_out, int out_size, void* d_ws, size_t ws_size,
                              hipStream_t stream) {
    (void)n_in; (void)out_size; (void)d_ws; (void)ws_size;
    const int B = in_sizes[0] / 9;          // states is (B, 9)
    dim3 grid(B / BPW), block(NTHREADS);
    frap_wmma_kernel<<<grid, block, 0, stream>>>(
        (const float*)d_in[0],  // states
        (const float*)d_in[1],  // p_w
        (const float*)d_in[2],  // d_w
        (const float*)d_in[3],  // d_b
        (const float*)d_in[4],  // lane_w
        (const float*)d_in[5],  // lane_b
        (const float*)d_in[6],  // lc_w
        (const float*)d_in[7],  // lc_b
        (const float*)d_in[8],  // rel_w
        (const float*)d_in[9],  // rc_w
        (const float*)d_in[10], // rc_b
        (const float*)d_in[11], // hid_w
        (const float*)d_in[12], // hid_b
        (const float*)d_in[13], // bm_w
        (const float*)d_in[14], // bm_b
        (const int*)d_in[15],   // phase_pairs
        (const int*)d_in[16],   // comp_mask
        (float*)d_out);
}
